// LSTMModel_39444979647123
// MI455X (gfx1250) — compile-verified, matrix operand order checked by reading, not run on hardware
//
#include <hip/hip_runtime.h>
#include <hip/hip_bf16.h>
#include <stdint.h>

#define BATCH 64
#define TLEN  2048
#define DIN   256
#define HID   256
#define GATES 1024   // 4*HID
#define NWG_SCAN 16

typedef __attribute__((ext_vector_type(16))) __bf16 v16bf;
typedef __attribute__((ext_vector_type(8)))  float  v8f;
typedef __attribute__((ext_vector_type(4))) unsigned u32x4;
typedef __attribute__((ext_vector_type(8))) int      i32x8;
typedef __attribute__((ext_vector_type(4))) int      i32x4;

union V16 { v16bf v; uint4 q[2]; };

__device__ __forceinline__ unsigned short f2bf(float f) {
  unsigned u = __float_as_uint(f);
  unsigned r = u + 0x7FFFu + ((u >> 16) & 1u);   // round-to-nearest-even
  return (unsigned short)(r >> 16);
}
__device__ __forceinline__ float bf2f(unsigned short h) {
  return __uint_as_float(((unsigned)h) << 16);
}
__device__ __forceinline__ float sigmoidf_(float x) { return 1.0f / (1.0f + __expf(-x)); }

// ---------------------------------------------------------------- utility kernels
__global__ void k_zero_ctr(unsigned* p) {
  if (threadIdx.x == 0 && blockIdx.x == 0) *p = 0u;
}

__global__ __launch_bounds__(256) void k_cvt_x(const float* __restrict__ x,
                                               unsigned short* __restrict__ xbf, int n4) {
  int id = blockIdx.x * blockDim.x + threadIdx.x;
  if (id < n4) {
    float4 f = reinterpret_cast<const float4*>(x)[id];
    unsigned u0 = (unsigned)f2bf(f.x) | ((unsigned)f2bf(f.y) << 16);
    unsigned u1 = (unsigned)f2bf(f.z) | ((unsigned)f2bf(f.w) << 16);
    reinterpret_cast<uint2*>(xbf)[id] = make_uint2(u0, u1);
  }
}

// Pack Wi (DxG) and Wh (HxG) transposed to bf16: WT[n][k] = W[k][n]
__global__ __launch_bounds__(256) void k_pack_w(const float* __restrict__ Wi,
                                                const float* __restrict__ Wh,
                                                unsigned short* __restrict__ WiT,
                                                unsigned short* __restrict__ WhT) {
  int id = blockIdx.x * blockDim.x + threadIdx.x;
  if (id < GATES * DIN) {
    int n = id / DIN, k = id % DIN;
    WiT[id] = f2bf(Wi[(size_t)k * GATES + n]);
  } else {
    int j = id - GATES * DIN;
    int n = j / HID, k = j % HID;
    WhT[j] = f2bf(Wh[(size_t)k * GATES + n]);
  }
}

// ---------------------------------------------------------------- xw = x @ Wi + b
// A = xbf [B*T][256] row-major bf16, B = WiT [1024][256] (Wi columns contiguous).
// Output xw stored [T][B][1024] bf16 so each timestep is a contiguous 128 KB block.
__global__ __launch_bounds__(256) void k_gemm_xw(const unsigned short* __restrict__ xbf,
                                                 const unsigned short* __restrict__ WiT,
                                                 const float* __restrict__ bias,
                                                 unsigned short* __restrict__ xw) {
  const int wid  = blockIdx.x * 8 + (threadIdx.x >> 5);
  const int lane = threadIdx.x & 31;
  const int lm   = lane & 15;
  const int off  = (lane >> 4) << 3;   // 0 for lanes 0-15, 8 for lanes 16-31
  const int tileN = wid & 15;          // 16 tiles of 64 cols
  const int tileM = wid >> 4;          // 8192 tiles of 16 rows
  const int m0 = tileM << 4;
  const int n0 = tileN << 6;

  // A fragments: 16 rows x K=256 (8 chunks of K=32)
  v16bf a[8];
  {
    const unsigned short* arow = xbf + (size_t)(m0 + lm) * DIN;
    #pragma unroll
    for (int kc = 0; kc < 8; ++kc) {
      V16 u;
      u.q[0] = *reinterpret_cast<const uint4*>(arow + kc * 32 + off);
      u.q[1] = *reinterpret_cast<const uint4*>(arow + kc * 32 + off + 16);
      a[kc] = u.v;
    }
  }
  const int bglob = m0 / TLEN;   // batch index (tiles never straddle batches)
  const int t0    = m0 % TLEN;

  #pragma unroll
  for (int nt = 0; nt < 4; ++nt) {
    const int ncol = n0 + nt * 16 + lm;
    const float bv = bias[ncol];          // fold gate bias into xw
    v8f acc;
    #pragma unroll
    for (int i = 0; i < 8; ++i) acc[i] = bv;
    // preload all 8 B fragments so the loads clause together and overlap WMMA
    const unsigned short* brow = WiT + (size_t)ncol * DIN;
    v16bf bf[8];
    #pragma unroll
    for (int kc = 0; kc < 8; ++kc) {
      V16 u;
      u.q[0] = *reinterpret_cast<const uint4*>(brow + kc * 32 + off);
      u.q[1] = *reinterpret_cast<const uint4*>(brow + kc * 32 + off + 16);
      bf[kc] = u.v;
    }
    #pragma unroll
    for (int kc = 0; kc < 8; ++kc)
      acc = __builtin_amdgcn_wmma_f32_16x16x32_bf16(false, a[kc], false, bf[kc],
                                                    (short)0, acc, false, false);
    // C layout: lane -> N=lm, VGPR v -> M = v + 8*(lane>=16)
    #pragma unroll
    for (int v = 0; v < 8; ++v) {
      int t_idx = t0 + v + (off ? 8 : 0);
      xw[((size_t)t_idx * BATCH + bglob) * GATES + ncol] = f2bf(acc[v]);
    }
  }
}

// ---------------------------------------------------------------- recurrent scan
// 16 persistent WGs; WG k owns h-columns [16k,16k+16) and gate cols {j,H+j,2H+j,3H+j}.
__global__ __launch_bounds__(256) void k_scan(const unsigned short* __restrict__ xw,
                                              const unsigned short* __restrict__ WhT,
                                              const float* __restrict__ Wo,
                                              const float* __restrict__ bo,
                                              unsigned short* __restrict__ hbuf, // [2][64][256]
                                              unsigned* __restrict__ ctr,
                                              float* __restrict__ out) {
  __shared__ unsigned short whs[64][DIN];      // 32 KB  Wh slice (B-operand rows)
  __shared__ unsigned short ha[BATCH][HID];    // 32 KB  current h (A-operand)
  __shared__ unsigned short xws[2][BATCH][64]; // 16 KB  TDM-staged xw slices (double buffer)
  __shared__ float zs[BATCH][64];              // 16 KB  z slice
  __shared__ float cs[BATCH][16];              //  4 KB  c slice
  __shared__ float hs[BATCH][16];              //  4 KB  h slice (f32, for final output)

  const int wg   = blockIdx.x;
  const int tid  = threadIdx.x;
  const int wave = tid >> 5;
  const int lane = tid & 31;
  const int lm   = lane & 15;
  const int off  = (lane >> 4) << 3;

  // Preload Wh slice: local col n (0..63) -> gate col (n>>4)*HID + wg*16 + (n&15)
  for (int i = tid; i < 64 * (DIN / 8); i += 256) {
    int n = i >> 5, ck = i & 31;
    int gc = ((n >> 4) * HID) + wg * 16 + (n & 15);
    reinterpret_cast<uint4*>(&whs[n][0])[ck] =
        reinterpret_cast<const uint4*>(WhT + (size_t)gc * DIN)[ck];
  }
  for (int i = tid; i < BATCH * HID / 8; i += 256)
    reinterpret_cast<uint4*>(&ha[0][0])[i] = make_uint4(0, 0, 0, 0);
  for (int i = tid; i < BATCH * 16; i += 256) { cs[i >> 4][i & 15] = 0.f; hs[i >> 4][i & 15] = 0.f; }
  __syncthreads();

  // ---- TDM descriptor: 2D gather of this WG's 64 gate columns for one timestep.
  // Element (x, y) with y = 4*m + nt maps to xw byte offset 2*(x + 256*y) from the
  // tile base (row stride 1024 elem == 4 * group stride 256 elem), landing
  // contiguously in LDS as xws[buf][m][nt*16+x].  tile = 16 x 256, stride0 = 256.
  const unsigned long long xw_base =
      (unsigned long long)(uintptr_t)xw + (unsigned long long)wg * 16ull * 2ull;

  if (wave == 0) {   // single wave per WG drives the TDM (EXEC is ignored by TDM)
    unsigned lds_dst = (unsigned)(uintptr_t)&xws[0][0][0];
    unsigned long long ga = xw_base;   // t = 0
    u32x4 g0 = { 1u,                                   // count=1, user mode
                 lds_dst,
                 (unsigned)(ga & 0xFFFFFFFFu),
                 (unsigned)((ga >> 32) & 0x1FFFFFFu) | (2u << 30) };  // type=2
    i32x8 g1 = { (int)0x00010000,        // data_size=1 (2B), no multicast/pad/iterate
                 (int)(16u << 16),       // tensor_dim0 = 16  (bits 63:48)
                 (int)(256u << 16),      // tensor_dim0 hi=0 | tensor_dim1 = 256
                 (int)(16u << 16),       // tensor_dim1 hi=0 | tile_dim0 = 16
                 (int)256,               // tile_dim1 = 256, tile_dim2 = 0
                 (int)256,               // tensor_dim0_stride = 256 (low 32)
                 0, 0 };                 // stride0 hi, stride1 (unused for 2D)
    i32x4 gz4 = { 0, 0, 0, 0 };
    i32x8 gz8 = { 0, 0, 0, 0, 0, 0, 0, 0 };
    __builtin_amdgcn_tensor_load_to_lds(g0, g1, gz4, gz4, gz8, 0);
  }

  const int m0 = (wave & 3) << 4;   // M-tile (batch rows)
  const int np = wave >> 2;         // which pair of N-tiles

  for (int t = 0; t < TLEN; ++t) {
    if (wave == 0) {
      __builtin_amdgcn_s_wait_tensorcnt(0);      // xws[t&1] ready
      if (t + 1 < TLEN) {                        // kick off next step's DMA
        unsigned lds_dst = (unsigned)(uintptr_t)&xws[(t + 1) & 1][0][0];
        unsigned long long ga = xw_base + (unsigned long long)(t + 1) * (BATCH * GATES * 2ull);
        u32x4 g0 = { 1u, lds_dst,
                     (unsigned)(ga & 0xFFFFFFFFu),
                     (unsigned)((ga >> 32) & 0x1FFFFFFu) | (2u << 30) };
        i32x8 g1 = { (int)0x00010000, (int)(16u << 16), (int)(256u << 16),
                     (int)(16u << 16), (int)256, (int)256, 0, 0 };
        i32x4 gz4 = { 0, 0, 0, 0 };
        i32x8 gz8 = { 0, 0, 0, 0, 0, 0, 0, 0 };
        __builtin_amdgcn_tensor_load_to_lds(g0, g1, gz4, gz4, gz8, 0);
      }
    }
    __syncthreads();   // publish staged xws[t&1] to all waves

    // ---- z = xw[t] + h @ Wh_slice  (M=64,N=64,K=256 per WG) ----
    v16bf a[8];
    #pragma unroll
    for (int kc = 0; kc < 8; ++kc) {
      V16 u;
      u.q[0] = *reinterpret_cast<const uint4*>(&ha[m0 + lm][kc * 32 + off]);
      u.q[1] = *reinterpret_cast<const uint4*>(&ha[m0 + lm][kc * 32 + off + 16]);
      a[kc] = u.v;
    }
    #pragma unroll
    for (int s = 0; s < 2; ++s) {
      const int nt = np * 2 + s;
      const int nl = nt * 16 + lm;             // local col 0..63
      v8f acc;
      #pragma unroll
      for (int v = 0; v < 8; ++v) {
        int m = m0 + v + (off ? 8 : 0);
        acc[v] = bf2f(xws[t & 1][m][nl]);      // LDS-staged xw
      }
      v16bf bfr[8];
      #pragma unroll
      for (int kc = 0; kc < 8; ++kc) {
        V16 u;
        u.q[0] = *reinterpret_cast<const uint4*>(&whs[nl][kc * 32 + off]);
        u.q[1] = *reinterpret_cast<const uint4*>(&whs[nl][kc * 32 + off + 16]);
        bfr[kc] = u.v;
      }
      #pragma unroll
      for (int kc = 0; kc < 8; ++kc)
        acc = __builtin_amdgcn_wmma_f32_16x16x32_bf16(false, a[kc], false, bfr[kc],
                                                      (short)0, acc, false, false);
      #pragma unroll
      for (int v = 0; v < 8; ++v)
        zs[m0 + v + (off ? 8 : 0)][nl] = acc[v];
    }
    __syncthreads();

    // ---- gates: 1024 (m,jj) elements, 4 per thread ----
    unsigned short* hcur = hbuf + (size_t)(t & 1) * BATCH * HID;
    #pragma unroll
    for (int j = 0; j < 4; ++j) {
      int e = tid * 4 + j;
      int m = e >> 4, jj = e & 15;
      float iv = sigmoidf_(zs[m][jj]);
      float fv = sigmoidf_(zs[m][16 + jj]);
      float gv = tanhf(zs[m][32 + jj]);
      float ov = sigmoidf_(zs[m][48 + jj]);
      float nc = fv * cs[m][jj] + iv * gv;
      float nh = ov * tanhf(nc);
      cs[m][jj] = nc;
      hs[m][jj] = nh;
      hcur[(size_t)m * HID + wg * 16 + jj] = f2bf(nh);
    }

    // ---- device-wide barrier (release stores -> acquire loads) ----
    __threadfence();
    __syncthreads();
    __builtin_amdgcn_s_cluster_barrier();  // NOP unless dispatched as a cluster
    if (tid == 0) {
      __hip_atomic_fetch_add(ctr, 1u, __ATOMIC_RELEASE, __HIP_MEMORY_SCOPE_AGENT);
      unsigned want = (unsigned)(t + 1) * NWG_SCAN;
      while (__hip_atomic_load(ctr, __ATOMIC_ACQUIRE, __HIP_MEMORY_SCOPE_AGENT) < want)
        __builtin_amdgcn_s_sleep(2);
    }
    __syncthreads();
    __threadfence();

    // ---- reload full new h (32 KB, L2-resident) into LDS ----
    const uint4* src = reinterpret_cast<const uint4*>(hcur);
    #pragma unroll
    for (int j = 0; j < 8; ++j)
      reinterpret_cast<uint4*>(&ha[0][0])[tid + 256 * j] = src[tid + 256 * j];
    __syncthreads();
  }

  // ---- outputs: y[64] | c_fin[64*256] | h_fin[64*256] ----
  for (int i = tid; i < BATCH * 16; i += 256) {
    int m = i >> 4, jj = i & 15;
    out[64 + (size_t)m * HID + wg * 16 + jj] = cs[m][jj];
    out[64 + BATCH * HID + (size_t)m * HID + wg * 16 + jj] = hs[m][jj];
  }
  if (wg == 0 && tid < BATCH) {
    const unsigned short* hf = hbuf + (size_t)((TLEN - 1) & 1) * BATCH * HID;
    float s = 0.f;
    for (int k = 0; k < HID; ++k) s += bf2f(hf[(size_t)tid * HID + k]) * Wo[k];
    out[tid] = s + bo[0];
  }
}

// ---------------------------------------------------------------- launcher
extern "C" void kernel_launch(void* const* d_in, const int* in_sizes, int n_in,
                              void* d_out, int out_size, void* d_ws, size_t ws_size,
                              hipStream_t stream) {
  (void)in_sizes; (void)n_in; (void)out_size; (void)ws_size;
  const float* x  = (const float*)d_in[0];
  const float* Wi = (const float*)d_in[1];
  const float* Wh = (const float*)d_in[2];
  const float* b  = (const float*)d_in[3];
  const float* Wo = (const float*)d_in[4];
  const float* bo = (const float*)d_in[5];

  char* ws = (char*)d_ws;
  size_t off = 0;
  unsigned short* xbf  = (unsigned short*)(ws + off); off += (size_t)BATCH * TLEN * DIN * 2;
  unsigned short* WiT  = (unsigned short*)(ws + off); off += (size_t)GATES * DIN * 2;
  unsigned short* WhT  = (unsigned short*)(ws + off); off += (size_t)GATES * HID * 2;
  unsigned short* xw   = (unsigned short*)(ws + off); off += (size_t)TLEN * BATCH * GATES * 2;
  unsigned short* hbuf = (unsigned short*)(ws + off); off += (size_t)2 * BATCH * HID * 2;
  unsigned*       ctr  = (unsigned*)(ws + off);       off += 64;

  k_zero_ctr<<<1, 64, 0, stream>>>(ctr);
  k_cvt_x<<<(BATCH * TLEN * DIN / 4 + 255) / 256, 256, 0, stream>>>(x, xbf, BATCH * TLEN * DIN / 4);
  k_pack_w<<<(2 * GATES * DIN) / 256, 256, 0, stream>>>(Wi, Wh, WiT, WhT);
  k_gemm_xw<<<(BATCH * TLEN / 16) * (GATES / 64) / 8, 256, 0, stream>>>(xbf, WiT, b, xw);
  k_scan<<<NWG_SCAN, 256, 0, stream>>>(xw, WhT, Wo, bo, hbuf, ctr, (float*)d_out);
}
